// ObservationModel_15118284882180
// MI455X (gfx1250) — matrix operand order repaired and checked by reading, not compiled, not run on hardware
//
#include <hip/hip_runtime.h>
#include <math.h>
#include <stdint.h>

// ---------------------------------------------------------------------------
// CDNA5 (gfx1250) types
// ---------------------------------------------------------------------------
typedef __attribute__((ext_vector_type(16))) __bf16 v16bf;
typedef __attribute__((ext_vector_type(8)))  __bf16 v8bf;
typedef __attribute__((ext_vector_type(8)))  float  v8f;
typedef __attribute__((ext_vector_type(4)))  unsigned int v4u;
typedef __attribute__((ext_vector_type(8)))  int v8i_t;
typedef __attribute__((ext_vector_type(4)))  int v4i_t;

#if defined(__AMDGCN__) && __has_builtin(__builtin_amdgcn_tensor_load_to_lds) && \
    __has_builtin(__builtin_amdgcn_s_wait_tensorcnt)
#define USE_TDM 1
#else
#define USE_TDM 0
#endif

// ---------------------------------------------------------------------------
// Generic bf16-WMMA GEMM:  C[M,N] = act(A[M,K] @ B[K,N] + bias[N]) (+ add[M,N])
// Block = 256 threads (8 waves).  Block tile: 16(M) x 128(N), K-step 64.
// Wave w owns the 16x16 C tile at n-offset w*16; 2 chained
// V_WMMA_F32_16X16X32_BF16 per K iteration.
// A tile is DMA'd global->LDS by the Tensor Data Mover (TENSOR_LOAD_TO_LDS,
// issued by wave 0, tracked by TENSORcnt) as raw f32 and converted to bf16 at
// fragment build; B tile is staged synchronously as bf16 with k-contiguous
// rows so each lane's B fragment is two 16-byte runs, 32 bytes apart (2x ds_load_b128).
// Requires M%16==0, N%128==0, K%64==0 (true for every layer here).
// ---------------------------------------------------------------------------
__global__ void __launch_bounds__(256)
gemm_bf16_wmma(const float* __restrict__ A, const float* __restrict__ B,
               const float* __restrict__ bias, const float* __restrict__ add,
               float* __restrict__ C, int M, int N, int K, int act)
{
    __shared__ float  Af32[16 * 64];   // A tile, f32, row-major [m][kt 0..63]
    __shared__ __bf16 Bs[128 * 72];    // B^T: row n -> kt 0..63 (+8 pad)

    const int tid  = threadIdx.x;
    const int lane = tid & 31;
    const int wave = tid >> 5;
    const int m0   = blockIdx.x * 16;
    const int n0   = blockIdx.y * 128;

    v8f acc = {};

    for (int k0 = 0; k0 < K; k0 += 64) {
        // ---- stage A (16 x 64 f32) ----
#if USE_TDM
        if (wave == 0) {
            // Tensor DMA descriptor (D#): 2D tile, dim0 = k (64 contiguous
            // elems), dim1 = m (16 rows), row stride = K elements.
            unsigned long long ga =
                (unsigned long long)(uintptr_t)(A + (size_t)m0 * K + k0);
            v4u g0 = {};
            g0[0] = 1u;                                    // count=1 (valid), user mode
            g0[1] = (unsigned)(uintptr_t)(void*)&Af32[0];  // lds_addr (LDS byte offset)
            g0[2] = (unsigned)(ga & 0xFFFFFFFFu);          // global_addr[31:0]
            g0[3] = (unsigned)((ga >> 32) & 0x01FFFFFFu) | (2u << 30); // addr[56:32], type=2
            v8i_t g1 = {};
            g1[0] = 0x20000;                               // data_size=2 (4 bytes)
            g1[1] = (int)(((unsigned)K & 0xFFFFu) << 16);  // tensor_dim0[15:0]
            g1[2] = (int)(((unsigned)K >> 16) | (((unsigned)M & 0xFFFFu) << 16));
            g1[3] = (int)(((unsigned)M >> 16) | (64u << 16));  // tile_dim0 = 64
            g1[4] = 16;                                    // tile_dim1 = 16, tile_dim2 = 0
            g1[5] = K;                                     // tensor_dim0_stride[31:0]
            v4i_t g2 = {}, g3 = {};
            v8i_t g4 = {};
            __builtin_amdgcn_tensor_load_to_lds(g0, g1, g2, g3, g4, 0);
        }
#else
        {   // synchronous fallback: one float4 per thread
            const int arow = tid & 15;
            const int ac   = (tid >> 4) * 4;
            *(float4*)&Af32[arow * 64 + ac] =
                *(const float4*)&A[(size_t)(m0 + arow) * K + (k0 + ac)];
        }
#endif
        // ---- stage B (64 x 128) -> Bs[n][kt] bf16, 2k x 4n units, unrolled ----
        #pragma unroll
        for (int u = 0; u < 4; ++u) {
            const int id = tid + u * 256;
            const int n  = (id & 31) * 4;
            const int kt = (id >> 5) * 2;
            const float4 b0 = *(const float4*)&B[(size_t)(k0 + kt)     * N + (n0 + n)];
            const float4 b1 = *(const float4*)&B[(size_t)(k0 + kt + 1) * N + (n0 + n)];
            union { __bf16 h[2]; unsigned int u32; } p0, p1, p2, p3;
            p0.h[0] = (__bf16)b0.x; p0.h[1] = (__bf16)b1.x;
            p1.h[0] = (__bf16)b0.y; p1.h[1] = (__bf16)b1.y;
            p2.h[0] = (__bf16)b0.z; p2.h[1] = (__bf16)b1.z;
            p3.h[0] = (__bf16)b0.w; p3.h[1] = (__bf16)b1.w;
            *(unsigned int*)&Bs[(n + 0) * 72 + kt] = p0.u32;
            *(unsigned int*)&Bs[(n + 1) * 72 + kt] = p1.u32;
            *(unsigned int*)&Bs[(n + 2) * 72 + kt] = p2.u32;
            *(unsigned int*)&Bs[(n + 3) * 72 + kt] = p3.u32;
        }
        if (k0 + 64 < K)   // prefetch next weight tile (global_prefetch_b8)
            __builtin_prefetch(&B[(size_t)(k0 + 64 + wave * 8) * N + n0 + lane * 4], 0, 1);
#if USE_TDM
        if (wave == 0) __builtin_amdgcn_s_wait_tensorcnt(0);
#endif
        __syncthreads();

        const int am = lane & 15;
        const int hi = lane >> 4;
        const int nn = wave * 16 + am;
        #pragma unroll
        for (int kb = 0; kb < 2; ++kb) {
            // A fragment: k = kb*32 + hi*8 + h (h<8) ; kb*32 + 16 + hi*8 + h-8
            const float* ar = &Af32[am * 64 + kb * 32 + hi * 8];
            const float4 a0 = *(const float4*)ar;
            const float4 a1 = *(const float4*)(ar + 4);
            const float4 a2 = *(const float4*)(ar + 16);
            const float4 a3 = *(const float4*)(ar + 20);
            v16bf av;
            av[0]  = (__bf16)a0.x; av[1]  = (__bf16)a0.y;
            av[2]  = (__bf16)a0.z; av[3]  = (__bf16)a0.w;
            av[4]  = (__bf16)a1.x; av[5]  = (__bf16)a1.y;
            av[6]  = (__bf16)a1.z; av[7]  = (__bf16)a1.w;
            av[8]  = (__bf16)a2.x; av[9]  = (__bf16)a2.y;
            av[10] = (__bf16)a2.z; av[11] = (__bf16)a2.w;
            av[12] = (__bf16)a3.x; av[13] = (__bf16)a3.y;
            av[14] = (__bf16)a3.z; av[15] = (__bf16)a3.w;
            const __bf16* bp = &Bs[nn * 72 + kb * 32 + hi * 8];
            v8bf blo = *(const v8bf*)bp;
            v8bf bhi = *(const v8bf*)(bp + 16);
            v16bf bv = __builtin_shufflevector(blo, bhi,
                        0,1,2,3,4,5,6,7,8,9,10,11,12,13,14,15);
            acc = __builtin_amdgcn_wmma_f32_16x16x32_bf16(
                      false, av, false, bv, (short)0, acc, false, false);
        }
        __syncthreads();
    }

    // Epilogue: C layout — VGPR r, lane l -> m = r + (l>>4)*8, n = l&15
    const int n     = n0 + wave * 16 + (lane & 15);
    const int mbase = m0 + (lane >> 4) * 8;
    const float bs  = bias[n];
    #pragma unroll
    for (int r = 0; r < 8; ++r) {
        int mm  = mbase + r;
        float v = acc[r] + bs;
        if (act == 1) v = fmaxf(v, 0.f);
        if (add) v += add[(size_t)mm * N + n];
        C[(size_t)mm * N + n] = v;
    }
}

// ---------------------------------------------------------------------------
// Time embedding: per batch b, 256 threads
// ---------------------------------------------------------------------------
__global__ void __launch_bounds__(256)
temb_kernel(const float* __restrict__ timep,
            const float* t1w, const float* t1b,
            const float* t2w, const float* t2b,
            const float* t3w, const float* t3b,
            float* __restrict__ emb)
{
    __shared__ float e1[256], e2[256];
    int b = blockIdx.x, j = threadIdx.x;
    float t = logf(timep[b]) * 0.25f;
    e1[j] = sinf(t * t1w[j] + t1b[j]);
    __syncthreads();
    float s = t2b[j];
    for (int i = 0; i < 256; ++i) s += e1[i] * t2w[i * 256 + j];
    e2[j] = sinf(s);
    __syncthreads();
    s = t3b[j];
    for (int i = 0; i < 256; ++i) s += e2[i] * t3w[i * 256 + j];
    emb[b * 256 + j] = s;
}

// ---------------------------------------------------------------------------
// Camera features: quat->R, cam_fts(14) -> sin MLP 14->128->128.  32 blocks.
// ---------------------------------------------------------------------------
__global__ void __launch_bounds__(128)
cam_kernel(const float* __restrict__ cpq, const float* __restrict__ intr,
           const float* c1w, const float* c1b, const float* c2w, const float* c2b,
           float* __restrict__ Rout, float* __restrict__ cout)
{
    __shared__ float fts[14];
    __shared__ float cc[128];
    int bc = blockIdx.x, j = threadIdx.x;
    const float* pq = cpq + bc * 7;
    if (j == 0) {
        float x = pq[3], y = pq[4], z = pq[5], w = pq[6];
        float R[9] = {1 - 2*(y*y + z*z), 2*(x*y - z*w),     2*(x*z + y*w),
                      2*(x*y + z*w),     1 - 2*(x*x + z*z), 2*(y*z - x*w),
                      2*(x*z - y*w),     2*(y*z + x*w),     1 - 2*(x*x + y*y)};
        fts[0] = pq[0]; fts[1] = pq[1]; fts[2] = pq[2];
        for (int i = 0; i < 9; ++i) { fts[3 + i] = R[i]; Rout[bc * 9 + i] = R[i]; }
        const float* it = intr + bc * 6;
        fts[12] = it[2] / it[1]; fts[13] = it[2] / it[1];
    }
    __syncthreads();
    float s = c1b[j];
    for (int i = 0; i < 14; ++i) s += fts[i] * c1w[i * 128 + j];
    cc[j] = sinf(s);
    __syncthreads();
    s = c2b[j];
    for (int i = 0; i < 128; ++i) s += cc[i] * c2w[i * 128 + j];
    cout[bc * 128 + j] = sinf(s);
}

// ---------------------------------------------------------------------------
// Projection + bilinear gather -> z_con rows (8192 x 384)
// row = ((b*32+o)*8+cd)*4 + cam ; [0:256]=img feature, [256:384]=cam feat
// ---------------------------------------------------------------------------
__global__ void __launch_bounds__(256)
gather_kernel(const float* __restrict__ dcc, const float* __restrict__ cpq,
              const float* __restrict__ intr, const float* __restrict__ Rr,
              const float* __restrict__ imgf, const float* __restrict__ cfeat,
              float* __restrict__ zcon)
{
    int r = blockIdx.x, f = threadIdx.x;
    int cam = r & 3, cd = (r >> 2) & 7, o = (r >> 5) & 31, b = r >> 10;
    const float* p  = dcc + ((size_t)((b * 32 + o) * 8 + cd)) * 3;
    const float* cp = cpq + (b * 4 + cam) * 7;
    const float* R  = Rr + (b * 4 + cam) * 9;
    const float* it = intr + (b * 4 + cam) * 6;
    float r0 = p[0] - cp[0], r1 = p[1] - cp[1], r2 = p[2] - cp[2];
    float pcx = R[0]*r0 + R[3]*r1 + R[6]*r2;       // R^T * rel
    float pcy = R[1]*r0 + R[4]*r1 + R[7]*r2;
    float pcz = R[2]*r0 + R[5]*r1 + R[8]*r2;
    float zc  = (fabsf(pcz) < 1e-6f) ? 1e-6f : pcz;
    float u = it[2] * pcx / zc + it[4];
    float v = it[3] * pcy / zc + it[5];
    float py = v / it[0] * 60.f;
    float px = u / it[1] * 80.f;
    int pi0 = (int)floorf(py), pj0 = (int)floorf(px);
    float offi = fminf(fmaxf(py - 0.5f, 0.f), 59.f);
    float offj = fminf(fmaxf(px - 0.5f, 0.f), 79.f);
    const float* base = imgf + ((size_t)(b * 4 + cam)) * 60 * 80 * 256;
    float val = 0.f, wsum = 0.f;
    #pragma unroll
    for (int s = 0; s < 4; ++s) {
        int si = s >> 1, sj = s & 1;
        float ri = fabsf(offi - (float)(pi0 + 1 - si)) + 0.01f;
        float rj = fabsf(offj - (float)(pj0 + 1 - sj)) + 0.01f;
        float w = ri * rj;
        wsum += w;
        int gi = pi0 + si; gi = gi < 0 ? 0 : (gi > 59 ? 59 : gi);
        int gj = pj0 + sj; gj = gj < 0 ? 0 : (gj > 79 ? 79 : gj);
        val += w * base[((size_t)gi * 80 + gj) * 256 + f];
    }
    zcon[(size_t)r * 384 + f] = val / wsum;
    if (f < 128) zcon[(size_t)r * 384 + 256 + f] = cfeat[(b * 4 + cam) * 128 + f];
}

// ---------------------------------------------------------------------------
// ag-stack: max-pool over NC(=4) + concat  (8192x256 -> 8192x512)
// ---------------------------------------------------------------------------
__global__ void __launch_bounds__(256)
poolcat_kernel(const float* __restrict__ x, float* __restrict__ xcat)
{
    int r = blockIdx.x, f = threadIdx.x;
    size_t base = (size_t)(r >> 2) * 4 * 256 + f;
    float m = fmaxf(fmaxf(x[base], x[base + 256]), fmaxf(x[base + 512], x[base + 768]));
    xcat[(size_t)r * 512 + f]       = x[(size_t)r * 256 + f];
    xcat[(size_t)r * 512 + 256 + f] = m;
}

// max over NC -> pooled (2048x256) and ctx = [pooled, emb] (2048x512)
__global__ void __launch_bounds__(256)
poolctx_kernel(const float* __restrict__ x, const float* __restrict__ emb,
               float* __restrict__ pooled, float* __restrict__ ctx)
{
    int g = blockIdx.x, f = threadIdx.x;
    size_t base = (size_t)g * 4 * 256 + f;
    float m = fmaxf(fmaxf(x[base], x[base + 256]), fmaxf(x[base + 512], x[base + 768]));
    pooled[(size_t)g * 256 + f] = m;
    ctx[(size_t)g * 512 + f] = m;
    int b = g / 256;   // g over (b, o, cd) with 32*8 = 256 per batch
    ctx[(size_t)g * 512 + 256 + f] = emb[b * 256 + f];
}

// sin(in[M,3] @ W[3,128] + b)
__global__ void __launch_bounds__(128)
sin3_kernel(const float* __restrict__ in, const float* w, const float* b,
            float* __restrict__ out)
{
    int row = blockIdx.x, j = threadIdx.x;
    float s = b[j];
    for (int i = 0; i < 3; ++i) s += in[row * 3 + i] * w[i * 128 + j];
    out[row * 128 + j] = sinf(s);
}

// ze[g, q'*32+c] = sum_q z[g, q*32+c] * zw[q, q']   (2048 x 512)
__global__ void __launch_bounds__(512)
ze_kernel(const float* __restrict__ z, const float* __restrict__ zw,
          float* __restrict__ ze)
{
    int g = blockIdx.x, t = threadIdx.x;
    int qp = t >> 5, c = t & 31;
    float s = 0.f;
    #pragma unroll
    for (int q = 0; q < 16; ++q) s += z[(size_t)g * 512 + q * 32 + c] * zw[q * 16 + qp];
    ze[(size_t)g * 512 + qp * 32 + c] = s;
}

// h = [pe(128 bcast over d), cen(128), ze(512), pooled(256)]  -> (2048 x 1024)
__global__ void __launch_bounds__(256)
hcat_kernel(const float* __restrict__ pe, const float* __restrict__ cen,
            const float* __restrict__ ze, const float* __restrict__ pooled,
            float* __restrict__ h)
{
    int g = blockIdx.x, f = threadIdx.x;
    int bo = g >> 3;
    if (f < 128) {
        h[(size_t)g * 1024 + f]       = pe[(size_t)bo * 128 + f];
        h[(size_t)g * 1024 + 128 + f] = cen[(size_t)g * 128 + f];
    }
    h[(size_t)g * 1024 + 256 + f] = ze[(size_t)g * 512 + f];
    h[(size_t)g * 1024 + 512 + f] = ze[(size_t)g * 512 + 256 + f];
    h[(size_t)g * 1024 + 768 + f] = pooled[(size_t)g * 256 + f];
}

// ---------------------------------------------------------------------------
// adaLN: out = LN(xin(+pre)) * ln_s + ln_b, modulated by e = EBUF[g] (512:
// scale | shift), optional relu, optional post-add.  In-place safe.
// ---------------------------------------------------------------------------
__global__ void __launch_bounds__(256)
adaln_kernel(const float* __restrict__ xin, const float* __restrict__ pre,
             const float* __restrict__ ebuf, const float* __restrict__ ln_s,
             const float* __restrict__ ln_b, float* __restrict__ out,
             int relu, const float* __restrict__ post)
{
    __shared__ float red[256];
    int g = blockIdx.x, f = threadIdx.x;
    float v = xin[(size_t)g * 256 + f];
    if (pre) v += pre[(size_t)g * 256 + f];
    red[f] = v; __syncthreads();
    for (int s = 128; s > 0; s >>= 1) { if (f < s) red[f] += red[f + s]; __syncthreads(); }
    float mean = red[0] * (1.f / 256.f); __syncthreads();
    float d = v - mean;
    red[f] = d * d; __syncthreads();
    for (int s = 128; s > 0; s >>= 1) { if (f < s) red[f] += red[f + s]; __syncthreads(); }
    float var = red[0] * (1.f / 256.f);
    float xn = d * rsqrtf(var + 1e-6f);
    xn = xn * ln_s[f] + ln_b[f];
    float r = xn * (1.f + ebuf[(size_t)g * 512 + f]) + ebuf[(size_t)g * 512 + 256 + f];
    if (relu) r = fmaxf(r, 0.f);
    if (post) r += post[(size_t)g * 256 + f];
    out[(size_t)g * 256 + f] = r;
}

// ---------------------------------------------------------------------------
// Tiny attention over nd=8 tokens, 8 heads x 128.  Block = (bo, head).
// ---------------------------------------------------------------------------
__global__ void __launch_bounds__(128)
attn_kernel(const float* __restrict__ q, const float* __restrict__ k,
            const float* __restrict__ v, float* __restrict__ o)
{
    __shared__ float qs[8][128], ks[8][128], vs[8][128], att[8][8];
    int blk = blockIdx.x;
    int head = blk & 7, bo = blk >> 3;
    int f = threadIdx.x;
    const float sc = 0.0883883476483184f;   // 1/sqrt(128)
    for (int t = 0; t < 8; ++t) {
        size_t idx = ((size_t)(bo * 8 + t)) * 1024 + head * 128 + f;
        qs[t][f] = q[idx] * sc; ks[t][f] = k[idx]; vs[t][f] = v[idx];
    }
    __syncthreads();
    if (f < 64) {
        int qt = f >> 3, kt = f & 7;
        float s = 0.f;
        for (int i = 0; i < 128; ++i) s += qs[qt][i] * ks[kt][i];
        att[qt][kt] = s;
    }
    __syncthreads();
    if (f < 8) {
        float m = -1e30f;
        for (int kt = 0; kt < 8; ++kt) m = fmaxf(m, att[f][kt]);
        float e[8], ssum = 0.f;
        for (int kt = 0; kt < 8; ++kt) { e[kt] = expf(att[f][kt] - m); ssum += e[kt]; }
        float inv = 1.f / ssum;
        for (int kt = 0; kt < 8; ++kt) att[f][kt] = e[kt] * inv;
    }
    __syncthreads();
    for (int qt = 0; qt < 8; ++qt) {
        float s = 0.f;
        for (int kt = 0; kt < 8; ++kt) s += att[qt][kt] * vs[kt][f];
        o[((size_t)(bo * 8 + qt)) * 1024 + head * 128 + f] = s;
    }
}

// out-layer: mean over 8 tokens + concat  (2048x256 -> 2048x512)
__global__ void __launch_bounds__(256)
meancat_kernel(const float* __restrict__ in, float* __restrict__ out)
{
    int r = blockIdx.x, f = threadIdx.x;
    int g0 = (r >> 3) << 3;
    float s = 0.f;
    for (int t = 0; t < 8; ++t) s += in[(size_t)(g0 + t) * 256 + f];
    s *= (1.f / 8.f);
    out[(size_t)r * 512 + f]       = in[(size_t)r * 256 + f];
    out[(size_t)r * 512 + 256 + f] = s;
}

// final mean over 8 tokens (2048x256 -> 256x256)
__global__ void __launch_bounds__(256)
meanrows_kernel(const float* __restrict__ in, float* __restrict__ out)
{
    int bo = blockIdx.x, f = threadIdx.x;
    float s = 0.f;
    for (int t = 0; t < 8; ++t) s += in[(size_t)(bo * 8 + t) * 256 + f];
    out[(size_t)bo * 256 + f] = s * (1.f / 8.f);
}

// out[bo] = fin[bo,:] . f2_w + f2_b
__global__ void __launch_bounds__(256)
f2_kernel(const float* __restrict__ fin, const float* __restrict__ w,
          const float* __restrict__ b, float* __restrict__ out)
{
    __shared__ float red[256];
    int bo = blockIdx.x, f = threadIdx.x;
    red[f] = fin[(size_t)bo * 256 + f] * w[f];
    __syncthreads();
    for (int s = 128; s > 0; s >>= 1) { if (f < s) red[f] += red[f + s]; __syncthreads(); }
    if (f == 0) out[bo] = red[0] + b[0];
}

// ---------------------------------------------------------------------------
// Host driver
// ---------------------------------------------------------------------------
extern "C" void kernel_launch(void* const* d_in, const int* in_sizes, int n_in,
                              void* d_out, int out_size, void* d_ws, size_t ws_size,
                              hipStream_t stream)
{
    (void)in_sizes; (void)n_in; (void)out_size; (void)ws_size;
    const float* const* in = (const float* const*)d_in;

    const float* pos  = in[0];
    const float* zin  = in[1];
    const float* dcc  = in[2];
    const float* cpq  = in[3];
    const float* intr = in[4];
    const float* imgf = in[5];
    const float* timep = in[6];

    int pi = 7;
    auto nxt = [&]() { return in[pi++]; };
    // dense params, dict order: (w,b) pairs
    const float *t1_w = nxt(), *t1_b = nxt(), *t2_w = nxt(), *t2_b = nxt();
    const float *t3_w = nxt(), *t3_b = nxt();
    const float *c1_w = nxt(), *c1_b = nxt(), *c2_w = nxt(), *c2_b = nxt();
    const float *agsk_w = nxt(), *agsk_b = nxt();
    const float *ag0a_w = nxt(), *ag0a_b = nxt(), *ag0b_w = nxt(), *ag0b_b = nxt();
    const float *ag1a_w = nxt(), *ag1a_b = nxt(), *ag1b_w = nxt(), *ag1b_b = nxt();
    const float *cen1_w = nxt(), *cen1_b = nxt(), *cen2_w = nxt(), *cen2_b = nxt();
    const float *pos1_w = nxt(), *pos1_b = nxt(), *pos2_w = nxt(), *pos2_b = nxt();
    const float *sk_w = nxt(), *sk_b = nxt();
    const float *l0_w = nxt(), *l0_b = nxt(), *l1_w = nxt(), *l1_b = nxt();
    const float *l2_w = nxt(), *l2_b = nxt();
    const float *o0_w = nxt(), *o0_b = nxt(), *o1_w = nxt(), *o1_b = nxt();
    const float *o2_w = nxt(), *o2_b = nxt();
    const float *f1_w = nxt(), *f1_b = nxt(), *f2_w = nxt(), *f2_b = nxt();
    const float *z_w = nxt();
    const float *atw[4][4], *atb[4][4];               // [m][q,k,v,o]
    for (int m = 0; m < 4; ++m)
        for (int t = 0; t < 4; ++t) { atw[m][t] = nxt(); atb[m][t] = nxt(); }
    // adaLN params: sk, l0..2, m0..3, o0..2 ; each w,b,ln_s,ln_b
    const float *alw[11], *alb[11], *als[11], *alB[11];
    for (int i = 0; i < 11; ++i) { alw[i] = nxt(); alb[i] = nxt(); als[i] = nxt(); alB[i] = nxt(); }
    enum { A_SK = 0, A_L0 = 1, A_L1 = 2, A_L2 = 3, A_M0 = 4, A_O0 = 8 };

    // workspace layout (floats)
    float* W = (float*)d_ws;
    size_t off = 0;
    auto alloc = [&](size_t n) { float* p = W + off; off += n; return p; };
    float* EMB    = alloc(8 * 256);
    float* RR     = alloc(32 * 9 + 64);
    float* CFEAT  = alloc(32 * 128);
    float* POOLED = alloc((size_t)2048 * 256);
    float* CTX    = alloc((size_t)2048 * 512);
    float* CENH   = alloc((size_t)2048 * 128);
    float* CEN    = alloc((size_t)2048 * 128);
    float* PEH    = alloc((size_t)256 * 128);
    float* PE     = alloc((size_t)256 * 128);
    float* ZEb    = alloc((size_t)2048 * 512);
    float* H      = alloc((size_t)2048 * 1024);
    float* HA     = alloc((size_t)2048 * 256);
    float* HB     = alloc((size_t)2048 * 256);
    float* SKIP2  = alloc((size_t)2048 * 256);
    float* EBUF   = alloc((size_t)2048 * 512);
    float* OUTCAT = alloc((size_t)2048 * 512);
    float* FIN    = alloc((size_t)256 * 256);
    float* BIG    = alloc((size_t)8192 * 512);   // xcat ; later q|k
    float* ZCON   = alloc((size_t)8192 * 384);   // z_con ; later v
    float* SKIPB  = alloc((size_t)8192 * 256);   // ag skip ; later attn o
    float* XA     = alloc((size_t)8192 * 256);
    float* QB = BIG;
    float* KB = BIG + (size_t)2048 * 1024;
    float* VB = ZCON;
    float* OB = SKIPB;

    auto gemm = [&](const float* A, const float* Bw, const float* bias,
                    const float* add, float* Cc, int M, int N, int K, int act) {
        dim3 grid(M / 16, N / 128);
        gemm_bf16_wmma<<<grid, dim3(256), 0, stream>>>(A, Bw, bias, add, Cc, M, N, K, act);
    };
    auto alngemm = [&](int idx) {
        gemm(CTX, alw[idx], alb[idx], nullptr, EBUF, 2048, 512, 512, 0);
    };
    auto adaln = [&](const float* xin, const float* pre, int idx, float* out,
                     int relu, const float* post) {
        adaln_kernel<<<dim3(2048), dim3(256), 0, stream>>>(
            xin, pre, EBUF, als[idx], alB[idx], out, relu, post);
    };

    // --- stage 1: embeddings / camera / gather ---------------------------------
    temb_kernel<<<dim3(8), dim3(256), 0, stream>>>(timep, t1_w, t1_b, t2_w, t2_b, t3_w, t3_b, EMB);
    cam_kernel<<<dim3(32), dim3(128), 0, stream>>>(cpq, intr, c1_w, c1_b, c2_w, c2_b, RR, CFEAT);
    gather_kernel<<<dim3(8192), dim3(256), 0, stream>>>(dcc, cpq, intr, RR, imgf, CFEAT, ZCON);

    // --- stage 2: ag aggregation stack (WMMA GEMMs, 8192 rows) -----------------
    gemm(ZCON, agsk_w, agsk_b, nullptr, SKIPB, 8192, 256, 384, 0);
    gemm(ZCON, ag0a_w, ag0a_b, SKIPB, XA, 8192, 256, 384, 1);      // relu(.)+skip
    poolcat_kernel<<<dim3(8192), dim3(256), 0, stream>>>(XA, BIG);
    gemm(BIG, ag0b_w, ag0b_b, nullptr, XA, 8192, 256, 512, 1);
    gemm(XA, ag1a_w, ag1a_b, SKIPB, ZCON, 8192, 256, 256, 1);
    poolcat_kernel<<<dim3(8192), dim3(256), 0, stream>>>(ZCON, BIG);
    gemm(BIG, ag1b_w, ag1b_b, nullptr, XA, 8192, 256, 512, 1);
    poolctx_kernel<<<dim3(2048), dim3(256), 0, stream>>>(XA, EMB, POOLED, CTX);

    // --- stage 3: cen / pe / ze / h concat -------------------------------------
    sin3_kernel<<<dim3(2048), dim3(128), 0, stream>>>(dcc, cen1_w, cen1_b, CENH);
    gemm(CENH, cen2_w, cen2_b, nullptr, CEN, 2048, 128, 128, 0);
    sin3_kernel<<<dim3(256), dim3(128), 0, stream>>>(pos, pos1_w, pos1_b, PEH);
    gemm(PEH, pos2_w, pos2_b, nullptr, PE, 256, 128, 128, 0);
    ze_kernel<<<dim3(2048), dim3(512), 0, stream>>>(zin, z_w, ZEb);
    hcat_kernel<<<dim3(2048), dim3(256), 0, stream>>>(PE, CEN, ZEb, POOLED, H);

    // --- stage 4: skip2 + l-layers (adaLN) -------------------------------------
    gemm(H, sk_w, sk_b, nullptr, HA, 2048, 256, 1024, 0);
    alngemm(A_SK); adaln(HA, nullptr, A_SK, SKIP2, 0, nullptr);
    gemm(H, l0_w, l0_b, nullptr, HA, 2048, 256, 1024, 0);
    alngemm(A_L0); adaln(HA, nullptr, A_L0, HB, 1, nullptr);
    gemm(HB, l1_w, l1_b, nullptr, HA, 2048, 256, 256, 0);
    alngemm(A_L1); adaln(HA, nullptr, A_L1, HB, 1, SKIP2);
    gemm(HB, l2_w, l2_b, nullptr, HA, 2048, 256, 256, 0);
    alngemm(A_L2); adaln(HA, nullptr, A_L2, HB, 1, nullptr);

    // --- stage 5: 4 attention blocks -------------------------------------------
    for (int m = 0; m < 4; ++m) {
        gemm(HB, atw[m][0], atb[m][0], nullptr, QB, 2048, 1024, 256, 0);
        gemm(HB, atw[m][1], atb[m][1], nullptr, KB, 2048, 1024, 256, 0);
        gemm(HB, atw[m][2], atb[m][2], nullptr, VB, 2048, 1024, 256, 0);
        attn_kernel<<<dim3(2048), dim3(128), 0, stream>>>(QB, KB, VB, OB);
        gemm(OB, atw[m][3], atb[m][3], nullptr, HA, 2048, 256, 1024, 0);
        alngemm(A_M0 + m);
        adaln(HB, HA, A_M0 + m, HB, 0, nullptr);   // adaln(h + o)
    }

    // --- stage 6: output head ---------------------------------------------------
    gemm(HB, o0_w, o0_b, nullptr, HA, 2048, 256, 256, 0);
    alngemm(A_O0 + 0); adaln(HA, nullptr, A_O0 + 0, HA, 1, nullptr);
    meancat_kernel<<<dim3(2048), dim3(256), 0, stream>>>(HA, OUTCAT);
    gemm(OUTCAT, o1_w, o1_b, nullptr, HA, 2048, 256, 512, 0);
    alngemm(A_O0 + 1); adaln(HA, nullptr, A_O0 + 1, HA, 1, nullptr);
    meancat_kernel<<<dim3(2048), dim3(256), 0, stream>>>(HA, OUTCAT);
    gemm(OUTCAT, o2_w, o2_b, nullptr, HA, 2048, 256, 512, 0);
    alngemm(A_O0 + 2); adaln(HA, nullptr, A_O0 + 2, HA, 1, nullptr);
    meancat_kernel<<<dim3(2048), dim3(256), 0, stream>>>(HA, OUTCAT);
    gemm(OUTCAT, f1_w, f1_b, nullptr, HA, 2048, 256, 512, 1);
    meanrows_kernel<<<dim3(256), dim3(256), 0, stream>>>(HA, FIN);
    f2_kernel<<<dim3(256), dim3(256), 0, stream>>>(FIN, f2_w, f2_b, (float*)d_out);
}
